// VectorQuantizer_8796093022594
// MI455X (gfx1250) — compile-verified, matrix operand order checked by reading, not run on hardware
//
#include <hip/hip_runtime.h>
#include <hip/hip_bf16.h>

#define K_CODES 4096
#define DIM     64
#define N_VEC   32768                 // 32 * 1024
#define N_ELEM  (N_VEC * DIM)         // 2097152

typedef __attribute__((ext_vector_type(16))) __bf16 v16bf;
typedef __attribute__((ext_vector_type(8)))  float  v8f;

union Frag {
    v16bf v;
    uint4 q[2];
};

// ---------------- workspace layout (byte offsets) ----------------
#define WS_ZBF   0                                   // N_ELEM * 2 bytes
#define WS_EBF   (WS_ZBF + N_ELEM * 2)               // K*DIM * 2
#define WS_ENORM (WS_EBF + K_CODES * DIM * 2)        // K * 4
#define WS_IDX   (WS_ENORM + K_CODES * 4)            // N_VEC * 4 (int)
#define WS_CNT   (WS_IDX + N_VEC * 4)                // K * 4     (accum, zeroed)
#define WS_DW    (WS_CNT + K_CODES * 4)              // K*DIM * 4 (accum, zeroed)
#define WS_NSUM  (WS_DW + K_CODES * DIM * 4)         // 4         (accum, zeroed)
#define WS_LOSS  (WS_NSUM + 4)                       // 4         (accum, zeroed)
#define N_ACCUM  (K_CODES + K_CODES * DIM + 2)       // floats zeroed each launch

// ---------------- output layout (float offsets) ----------------
#define OUT_ZQ    0
#define OUT_IDX   (OUT_ZQ + N_ELEM)
#define OUT_LOSS  (OUT_IDX + N_VEC)
#define OUT_EMB   (OUT_LOSS + 1)
#define OUT_CS    (OUT_EMB + K_CODES * DIM)
#define OUT_EMAW  (OUT_CS + K_CODES)

__device__ __forceinline__ unsigned short f2bf_rne(float f) {
    unsigned u = __float_as_uint(f);
    u += 0x7FFFu + ((u >> 16) & 1u);   // round to nearest even
    return (unsigned short)(u >> 16);
}

// ---- convert fp32 -> bf16 (generic) ----
__global__ void k_cvt_bf16(const float* __restrict__ src,
                           unsigned short* __restrict__ dst, int n) {
    int i = blockIdx.x * blockDim.x + threadIdx.x;
    if (i < n) dst[i] = f2bf_rne(src[i]);
}

// ---- exact fp32 ||e_k||^2 ----
__global__ void k_enorm(const float* __restrict__ emb, float* __restrict__ enorm) {
    int k = blockIdx.x * blockDim.x + threadIdx.x;   // 4096 threads
    float s = 0.f;
    const float* row = emb + k * DIM;
    #pragma unroll 8
    for (int d = 0; d < DIM; ++d) { float v = row[d]; s += v * v; }
    enorm[k] = s;
}

// ---- zero accumulators ----
__global__ void k_zero(float* __restrict__ p, int n) {
    int i = blockIdx.x * blockDim.x + threadIdx.x;
    if (i < n) p[i] = 0.f;
}

// ---- WMMA distance + argmin: one wave per 16-row z tile ----
__global__ void k_argmin(const unsigned short* __restrict__ zbf,
                         const unsigned short* __restrict__ ebf,
                         const float* __restrict__ enorm,
                         int* __restrict__ idx_out,
                         float* __restrict__ idxf_out) {
    const int lane = threadIdx.x & 31;
    const int wave = threadIdx.x >> 5;
    const int rowbase = blockIdx.x * 128 + wave * 16;   // 8 waves * 16 rows
    const int lm = lane & 15;
    const int hi = lane >> 4;

    // A fragments (z tile, 16x64 bf16 as 2x 16x32). ISA 16-bit A layout:
    // lane half 'hi' picks K-subsets {0-7,16-23} (hi=0) / {8-15,24-31} (hi=1),
    // i.e. two contiguous 16B runs per fragment.
    Frag a0, a1;
    {
        const unsigned short* zr = zbf + (rowbase + lm) * DIM;
        a0.q[0] = *(const uint4*)(zr +  0 + hi * 8);
        a0.q[1] = *(const uint4*)(zr + 16 + hi * 8);
        a1.q[0] = *(const uint4*)(zr + 32 + hi * 8);
        a1.q[1] = *(const uint4*)(zr + 48 + hi * 8);
    }

    float minv[8];
    int   mini[8];
    #pragma unroll
    for (int r = 0; r < 8; ++r) { minv[r] = 3.4e38f; mini[r] = 0; }

    for (int ct = 0; ct < K_CODES / 16; ++ct) {
        const unsigned short* er = ebf + (ct * 16 + lm) * DIM;
        __builtin_prefetch(er + 16 * DIM, 0, 1);        // next column tile
        Frag b0, b1;                                    // B: col n in lane, same K packing
        b0.q[0] = *(const uint4*)(er +  0 + hi * 8);
        b0.q[1] = *(const uint4*)(er + 16 + hi * 8);
        b1.q[0] = *(const uint4*)(er + 32 + hi * 8);
        b1.q[1] = *(const uint4*)(er + 48 + hi * 8);

        v8f acc = {};
        acc = __builtin_amdgcn_wmma_f32_16x16x32_bf16(false, a0.v, false, b0.v,
                                                      (short)0, acc, false, false);
        acc = __builtin_amdgcn_wmma_f32_16x16x32_bf16(false, a1.v, false, b1.v,
                                                      (short)0, acc, false, false);

        const float en   = enorm[ct * 16 + lm];
        const int   cand = ct * 16 + lm;
        #pragma unroll
        for (int r = 0; r < 8; ++r) {
            float d = en - 2.0f * acc[r];               // ||e||^2 - 2 z.e  (argmin-equivalent)
            if (d < minv[r]) { minv[r] = d; mini[r] = cand; }
        }
    }

    // reduce (min, argmin) across the 16 lanes of each half-wave
    #pragma unroll
    for (int r = 0; r < 8; ++r) {
        float v = minv[r];
        int   i = mini[r];
        #pragma unroll
        for (int off = 8; off >= 1; off >>= 1) {
            float ov = __shfl_xor(v, off, 32);
            int   oi = __shfl_xor(i, off, 32);
            if (ov < v || (ov == v && oi < i)) { v = ov; i = oi; }
        }
        if (lm == 0) {                                  // C layout: lanes 0-15 -> M=r, 16-31 -> M=8+r
            int row = rowbase + hi * 8 + r;
            idx_out[row]  = i;
            idxf_out[row] = (float)i;
        }
    }
}

// ---- gather z_q, loss partials, counts/dw scatter ----
__global__ void k_gather(const float* __restrict__ z,
                         const float* __restrict__ emb,
                         const int* __restrict__ idx,
                         float* __restrict__ zq_out,
                         float* __restrict__ dw,
                         float* __restrict__ counts,
                         float* __restrict__ loss_sum) {
    __shared__ float red[256];
    int gid = blockIdx.x * 256 + threadIdx.x;           // N_ELEM threads
    int n = gid >> 6, d = gid & 63;
    int k = idx[n];
    float zv = z[gid];
    float q  = emb[k * DIM + d];
    zq_out[gid] = q;                                    // z_q_st == z_q numerically
    float diff = zv - q;
    unsafeAtomicAdd(&dw[k * DIM + d], zv);
    if (d == 0) unsafeAtomicAdd(&counts[k], 1.0f);
    red[threadIdx.x] = diff * diff;
    __syncthreads();
    for (int s = 128; s > 0; s >>= 1) {
        if (threadIdx.x < s) red[threadIdx.x] += red[threadIdx.x + s];
        __syncthreads();
    }
    if (threadIdx.x == 0) unsafeAtomicAdd(loss_sum, red[0]);
}

// ---- EMA blends + sum(new_cluster_size) ----
__global__ void k_fin1(const float* __restrict__ ema_w,
                       const float* __restrict__ ecs,
                       const float* __restrict__ counts,
                       const float* __restrict__ dw,
                       float* __restrict__ out_emaw,
                       float* __restrict__ out_cs,
                       float* __restrict__ nsum) {
    int t = blockIdx.x * 256 + threadIdx.x;             // K*DIM threads
    int k = t >> 6, d = t & 63;
    out_emaw[t] = 0.99f * ema_w[t] + 0.01f * dw[t];
    if (d == 0) {
        float ncs = 0.99f * ecs[k] + 0.01f * counts[k];
        out_cs[k] = ncs;
        unsafeAtomicAdd(nsum, ncs);
    }
}

// ---- normalized embedding + loss scalar ----
__global__ void k_fin2(const float* __restrict__ out_emaw,
                       const float* __restrict__ out_cs,
                       const float* __restrict__ nsum,
                       const float* __restrict__ loss_sum,
                       float* __restrict__ out_emb,
                       float* __restrict__ out_loss) {
    int t = blockIdx.x * 256 + threadIdx.x;             // K*DIM threads
    int k = t >> 6;
    float n  = *nsum;
    float cs = (out_cs[k] + 1e-5f) / (n + K_CODES * 1e-5f) * n;
    out_emb[t] = out_emaw[t] / cs;
    if (t == 0) *out_loss = 0.25f * (*loss_sum) / (float)N_ELEM;
}

extern "C" void kernel_launch(void* const* d_in, const int* in_sizes, int n_in,
                              void* d_out, int out_size, void* d_ws, size_t ws_size,
                              hipStream_t stream) {
    const float* z     = (const float*)d_in[0];
    const float* emb   = (const float*)d_in[1];
    const float* ecs   = (const float*)d_in[2];
    const float* ema_w = (const float*)d_in[3];

    char* ws = (char*)d_ws;
    unsigned short* zbf   = (unsigned short*)(ws + WS_ZBF);
    unsigned short* ebf   = (unsigned short*)(ws + WS_EBF);
    float*          enorm = (float*)(ws + WS_ENORM);
    int*            idx   = (int*)(ws + WS_IDX);
    float*          cnt   = (float*)(ws + WS_CNT);
    float*          dw    = (float*)(ws + WS_DW);
    float*          nsum  = (float*)(ws + WS_NSUM);
    float*          lsum  = (float*)(ws + WS_LOSS);

    float* out = (float*)d_out;
    float* out_zq   = out + OUT_ZQ;
    float* out_idx  = out + OUT_IDX;
    float* out_loss = out + OUT_LOSS;
    float* out_emb  = out + OUT_EMB;
    float* out_cs   = out + OUT_CS;
    float* out_emaw = out + OUT_EMAW;

    // 1) prep: bf16 conversions, ||e||^2, zero accumulators
    k_cvt_bf16<<<N_ELEM / 256, 256, 0, stream>>>(z, zbf, N_ELEM);
    k_cvt_bf16<<<(K_CODES * DIM) / 256, 256, 0, stream>>>(emb, ebf, K_CODES * DIM);
    k_enorm<<<K_CODES / 256, 256, 0, stream>>>(emb, enorm);
    k_zero<<<(N_ACCUM + 255) / 256, 256, 0, stream>>>(cnt, N_ACCUM);

    // 2) WMMA distance + argmin (256 blocks * 8 waves * 16 rows = 32768 rows)
    k_argmin<<<N_VEC / 128, 256, 0, stream>>>(zbf, ebf, enorm, idx, out_idx);

    // 3) gather z_q, loss, counts/dw
    k_gather<<<N_ELEM / 256, 256, 0, stream>>>(z, emb, idx, out_zq, dw, cnt, lsum);

    // 4) EMA blends + n
    k_fin1<<<(K_CODES * DIM) / 256, 256, 0, stream>>>(ema_w, ecs, cnt, dw,
                                                      out_emaw, out_cs, nsum);

    // 5) normalized embedding + loss
    k_fin2<<<(K_CODES * DIM) / 256, 256, 0, stream>>>(out_emaw, out_cs, nsum, lsum,
                                                      out_emb, out_loss);
}
